// ShortConv_68470368633596
// MI455X (gfx1250) — matrix operand order repaired and checked by reading, not compiled, not run on hardware
//
#include <hip/hip_runtime.h>
#include <hip/hip_bf16.h>
#include <stdint.h>

typedef __bf16 bf16;
typedef __attribute__((ext_vector_type(16))) __bf16 v16bf;
typedef __attribute__((ext_vector_type(8)))  __bf16 v8bf;
typedef __attribute__((ext_vector_type(4)))  __bf16 v4bf;
typedef __attribute__((ext_vector_type(8)))  float  v8f;
typedef __attribute__((ext_vector_type(4)))  float  v4f;
typedef __attribute__((ext_vector_type(4)))  int    v4i;

#define D_DIM   1024
#define L_DIM   2048
#define B_DIM   16
#define M_DIM   (B_DIM * L_DIM)   // 32768
#define BM      128
#define BN      64
#define KC      32
#define NKITER  (D_DIM / KC)      // 32

// ---- gfx1250 async global->LDS path (ASYNCcnt), guarded so compile never breaks
#if __has_builtin(__builtin_amdgcn_global_load_async_to_lds_b128) && \
    __has_builtin(__builtin_amdgcn_s_wait_asynccnt)
#define HAVE_ASYNC 1
#else
#define HAVE_ASYNC 0
#endif

typedef __attribute__((address_space(1))) v4i glob_v4i;   // global AS
typedef __attribute__((address_space(3))) v4i lds_v4i;    // LDS AS

static __device__ __forceinline__ void cp16(bf16* dst_lds, const bf16* src_glb) {
#if HAVE_ASYNC
  __builtin_amdgcn_global_load_async_to_lds_b128(
      (glob_v4i*)(void*)src_glb,
      (lds_v4i*)(void*)dst_lds, 0, 0);
#else
  *(v8bf*)dst_lds = *(const v8bf*)src_glb;
#endif
}

#if HAVE_ASYNC
#define WAIT_ASYNC(n) __builtin_amdgcn_s_wait_asynccnt(n)
#else
#define WAIT_ASYNC(n) ((void)0)
#endif

static __device__ __forceinline__ v16bf cat8(v8bf a, v8bf b) {
  return __builtin_shufflevector(a, b, 0,1,2,3,4,5,6,7,8,9,10,11,12,13,14,15);
}

// ---------------------------------------------------------------- fp32 -> bf16
__global__ __launch_bounds__(256) void cvt_f32_bf16_kernel(
    const float* __restrict__ src, bf16* __restrict__ dst, int n4) {
  int i = blockIdx.x * blockDim.x + threadIdx.x;
  if (i < n4) {
    v4f f = ((const v4f*)src)[i];
    v4bf o;
    o[0] = (bf16)f[0]; o[1] = (bf16)f[1]; o[2] = (bf16)f[2]; o[3] = (bf16)f[3];
    ((v4bf*)dst)[i] = o;
  }
}

// ------------------------------------------------- GEMM1 + gate fusion (bf16)
// Bx = (Bg+binB)*(xg+binx), Cg = Cg+binC   (both bf16, row-major MxD)
__global__ __launch_bounds__(256) void gemm1_gate_kernel(
    const bf16*  __restrict__ xB,     // (M, D) bf16
    const bf16*  __restrict__ WinB,   // (3D, D) bf16 (N-major, K-contig)
    const float* __restrict__ bin,    // (3D)
    bf16* __restrict__ Bx,            // (M, D) bf16
    bf16* __restrict__ Cg)            // (M, D) bf16
{
  // double-buffered: [A 128x32 | B 3x64x32] x2 = 40 KB
  __shared__ __align__(32) bf16 smem[2 * (BM * KC + 3 * BN * KC)];
  const int bufStride = BM * KC + 3 * BN * KC;

  const int m0 = blockIdx.x * BM;
  const int n0 = blockIdx.y * BN;
  const int t    = threadIdx.x;
  const int lane = t & 31;
  const int wid  = t >> 5;
  const int wm   = wid & 3;    // 4 waves over M
  const int wn   = wid >> 2;   // 2 waves over N
  const int lr   = lane & 15;
  const int lh   = lane >> 4;

  v8f acc[3][2][2] = {};       // [gate][row-tile][col-tile]

  auto stage = [&](int kb, int k0) {
    bf16* sA = smem + kb * bufStride;
    bf16* sB = sA + BM * KC;
    {  // A: 8 KB, 32 B per thread
      const int row = t >> 1, kq = (t & 1) * 16;
      const bf16* s = xB + (size_t)(m0 + row) * D_DIM + k0 + kq;
      bf16* d = sA + row * KC + kq;
      cp16(d, s);
      cp16(d + 8, s + 8);
    }
    {  // B: 3 gates x 64x32, 16 B per thread per gate
      const int row = t >> 2, kq = (t & 3) * 8;
      #pragma unroll
      for (int g = 0; g < 3; ++g)
        cp16(sB + g * BN * KC + row * KC + kq,
             WinB + (size_t)(g * D_DIM + n0 + row) * D_DIM + k0 + kq);
    }
  };

  stage(0, 0);
  for (int kc = 0; kc < NKITER; ++kc) {
    const int cur = kc & 1;
    if (kc + 1 < NKITER) { stage(cur ^ 1, (kc + 1) * KC); WAIT_ASYNC(5); }
    else                 { WAIT_ASYNC(0); }
    __syncthreads();

    const bf16* sA = smem + cur * bufStride;
    const bf16* sB = sA + BM * KC;

    v16bf afrag[2];
    #pragma unroll
    for (int rt = 0; rt < 2; ++rt) {
      const bf16* p = sA + (wm * 32 + rt * 16 + lr) * KC + lh * 16;
      afrag[rt] = cat8(*(const v8bf*)p, *(const v8bf*)(p + 8));
    }
    v16bf bfrag[2][3];
    #pragma unroll
    for (int ct = 0; ct < 2; ++ct)
      #pragma unroll
      for (int g = 0; g < 3; ++g)
        bfrag[ct][g] = *(const v16bf*)(sB + g * BN * KC +
                                       (wn * 32 + ct * 16 + lr) * KC + lh * 16);
    #pragma unroll
    for (int g = 0; g < 3; ++g)
      #pragma unroll
      for (int ct = 0; ct < 2; ++ct)
        #pragma unroll
        for (int rt = 0; rt < 2; ++rt)
          acc[g][rt][ct] = __builtin_amdgcn_wmma_f32_16x16x32_bf16(
              false, afrag[rt], false, bfrag[ct][g], (short)0, acc[g][rt][ct],
              false, false);
    __syncthreads();
  }

  // ---- epilogue: bias + gate combine, repack via LDS, coalesced 16B stores
  float bB[2], bC[2], bx[2];
  #pragma unroll
  for (int ct = 0; ct < 2; ++ct) {
    const int c = n0 + wn * 32 + ct * 16 + lr;
    bB[ct] = bin[c];
    bC[ct] = bin[D_DIM + c];
    bx[ct] = bin[2 * D_DIM + c];
  }
  bf16* sOut = smem;  // 128x64 bf16 = 16 KB scratch
  #pragma unroll
  for (int sel = 0; sel < 2; ++sel) {
    __syncthreads();
    #pragma unroll
    for (int ct = 0; ct < 2; ++ct) {
      const int c = wn * 32 + ct * 16 + lr;
      #pragma unroll
      for (int rt = 0; rt < 2; ++rt)
        #pragma unroll
        for (int v = 0; v < 8; ++v) {
          const int r = wm * 32 + rt * 16 + v + lh * 8;
          const float val = (sel == 0)
              ? (acc[0][rt][ct][v] + bB[ct]) * (acc[2][rt][ct][v] + bx[ct])
              : (acc[1][rt][ct][v] + bC[ct]);
          sOut[r * BN + c] = (bf16)val;
        }
    }
    __syncthreads();
    {
      const int row = t >> 1, off = (t & 1) * 32;
      bf16* gd = (sel ? Cg : Bx) + (size_t)(m0 + row) * D_DIM + n0 + off;
      const bf16* ss = sOut + row * BN + off;
      #pragma unroll
      for (int i = 0; i < 32; i += 8)
        *(v8bf*)(gd + i) = *(const v8bf*)(ss + i);
    }
  }
}

// --------------------------------------- causal depthwise conv (K=3) + Cg mul
// y[l,d] = Cg[l,d] * (w0*Bx[l-2,d] + w1*Bx[l-1,d] + w2*Bx[l,d]);  in-place Cg->y
__global__ __launch_bounds__(256) void conv_gate_kernel(
    const bf16* __restrict__ Bx, const float* __restrict__ wconv,
    bf16* __restrict__ CgY) {
  const int i   = blockIdx.x * blockDim.x + threadIdx.x;  // 8-wide group index
  const int row = i >> 7;           // D/8 = 128 groups per row
  const int gp  = i & 127;
  const int d   = gp * 8;
  const int l   = row & (L_DIM - 1);

  const v8bf z = {};
  const v8bf* bx = (const v8bf*)Bx;
  v8bf b0  = bx[(size_t)row * 128 + gp];
  v8bf bm1 = (l >= 1) ? bx[(size_t)(row - 1) * 128 + gp] : z;
  v8bf bm2 = (l >= 2) ? bx[(size_t)(row - 2) * 128 + gp] : z;
  v8bf cg  = ((const v8bf*)CgY)[(size_t)row * 128 + gp];

  float wf[24];
  const v4f* wv = (const v4f*)(wconv + d * 3);  // 24 floats, 16B aligned
  #pragma unroll
  for (int q = 0; q < 6; ++q) {
    v4f f = wv[q];
    wf[q*4+0] = f[0]; wf[q*4+1] = f[1]; wf[q*4+2] = f[2]; wf[q*4+3] = f[3];
  }
  v8bf y;
  #pragma unroll
  for (int s = 0; s < 8; ++s) {
    const float conv = wf[s*3+0] * (float)bm2[s] + wf[s*3+1] * (float)bm1[s] +
                       wf[s*3+2] * (float)b0[s];
    y[s] = (bf16)((float)cg[s] * conv);
  }
  ((v8bf*)CgY)[(size_t)row * 128 + gp] = y;
}

// ---------------------------------------------------- GEMM2: out = y@Wout^T+b
__global__ __launch_bounds__(256) void gemm2_kernel(
    const bf16*  __restrict__ y,      // (M, D) bf16
    const bf16*  __restrict__ WoutB,  // (D, D) bf16 (N-major, K-contig)
    const float* __restrict__ bout,   // (D)
    float* __restrict__ out)          // (M, D) f32
{
  __shared__ __align__(32) bf16 smem[2 * (BM * KC + BN * KC)];  // 24 KB
  const int bufStride = BM * KC + BN * KC;

  const int m0 = blockIdx.x * BM;
  const int n0 = blockIdx.y * BN;
  const int t    = threadIdx.x;
  const int lane = t & 31;
  const int wid  = t >> 5;
  const int wm   = wid & 3;
  const int wn   = wid >> 2;
  const int lr   = lane & 15;
  const int lh   = lane >> 4;

  v8f acc[2][2] = {};

  auto stage = [&](int kb, int k0) {
    bf16* sA = smem + kb * bufStride;
    bf16* sB = sA + BM * KC;
    {
      const int row = t >> 1, kq = (t & 1) * 16;
      const bf16* s = y + (size_t)(m0 + row) * D_DIM + k0 + kq;
      bf16* d = sA + row * KC + kq;
      cp16(d, s);
      cp16(d + 8, s + 8);
    }
    {
      const int row = t >> 2, kq = (t & 3) * 8;
      cp16(sB + row * KC + kq,
           WoutB + (size_t)(n0 + row) * D_DIM + k0 + kq);
    }
  };

  stage(0, 0);
  for (int kc = 0; kc < NKITER; ++kc) {
    const int cur = kc & 1;
    if (kc + 1 < NKITER) { stage(cur ^ 1, (kc + 1) * KC); WAIT_ASYNC(3); }
    else                 { WAIT_ASYNC(0); }
    __syncthreads();

    const bf16* sA = smem + cur * bufStride;
    const bf16* sB = sA + BM * KC;

    v16bf afrag[2];
    #pragma unroll
    for (int rt = 0; rt < 2; ++rt) {
      const bf16* p = sA + (wm * 32 + rt * 16 + lr) * KC + lh * 16;
      afrag[rt] = cat8(*(const v8bf*)p, *(const v8bf*)(p + 8));
    }
    v16bf bfrag[2];
    #pragma unroll
    for (int ct = 0; ct < 2; ++ct)
      bfrag[ct] = *(const v16bf*)(sB + (wn * 32 + ct * 16 + lr) * KC + lh * 16);
    #pragma unroll
    for (int ct = 0; ct < 2; ++ct)
      #pragma unroll
      for (int rt = 0; rt < 2; ++rt)
        acc[rt][ct] = __builtin_amdgcn_wmma_f32_16x16x32_bf16(
            false, afrag[rt], false, bfrag[ct], (short)0, acc[rt][ct],
            false, false);
    __syncthreads();
  }

  #pragma unroll
  for (int ct = 0; ct < 2; ++ct) {
    const int c = n0 + wn * 32 + ct * 16 + lr;
    const float bb = bout[c];
    #pragma unroll
    for (int rt = 0; rt < 2; ++rt)
      #pragma unroll
      for (int v = 0; v < 8; ++v) {
        const int row = m0 + wm * 32 + rt * 16 + v + lh * 8;
        out[(size_t)row * D_DIM + c] = acc[rt][ct][v] + bb;
      }
  }
}

// ----------------------------------------------------------------- launcher
extern "C" void kernel_launch(void* const* d_in, const int* in_sizes, int n_in,
                              void* d_out, int out_size, void* d_ws, size_t ws_size,
                              hipStream_t stream) {
  const float* x     = (const float*)d_in[0];
  const float* Win   = (const float*)d_in[1];
  const float* bin   = (const float*)d_in[2];
  const float* wconv = (const float*)d_in[3];
  const float* Wout  = (const float*)d_in[4];
  const float* bout  = (const float*)d_in[5];
  float* out = (float*)d_out;

  char* ws = (char*)d_ws;
  bf16* WinB  = (bf16*)ws;  ws += (size_t)3 * D_DIM * D_DIM * sizeof(bf16);
  bf16* WoutB = (bf16*)ws;  ws += (size_t)D_DIM * D_DIM * sizeof(bf16);
  bf16* xBuf  = (bf16*)ws;  ws += (size_t)M_DIM * D_DIM * sizeof(bf16);
  bf16* Bx    = (bf16*)ws;  ws += (size_t)M_DIM * D_DIM * sizeof(bf16);
  bf16* CgY   = (bf16*)ws;  ws += (size_t)M_DIM * D_DIM * sizeof(bf16);
  if ((size_t)(ws - (char*)d_ws) > ws_size) return;  // scratch too small

  cvt_f32_bf16_kernel<<<(3 * D_DIM * D_DIM / 4) / 256, 256, 0, stream>>>(
      Win, WinB, 3 * D_DIM * D_DIM / 4);
  cvt_f32_bf16_kernel<<<(D_DIM * D_DIM / 4) / 256, 256, 0, stream>>>(
      Wout, WoutB, D_DIM * D_DIM / 4);
  cvt_f32_bf16_kernel<<<((size_t)M_DIM * D_DIM / 4) / 256, 256, 0, stream>>>(
      x, xBuf, M_DIM * (D_DIM / 4));

  dim3 g1(M_DIM / BM, D_DIM / BN);  // 256 x 16
  gemm1_gate_kernel<<<g1, 256, 0, stream>>>(xBuf, WinB, bin, Bx, CgY);

  conv_gate_kernel<<<(M_DIM * (D_DIM / 8)) / 256, 256, 0, stream>>>(Bx, wconv, CgY);

  dim3 g2(M_DIM / BM, D_DIM / BN);
  gemm2_kernel<<<g2, 256, 0, stream>>>(CgY, WoutB, bout, out);
}
